// QuantumHybridTransformerClassifier_65481071395390
// MI455X (gfx1250) — matrix operand order repaired and checked by reading, compiled, not run on hardware
//
#include <hip/hip_runtime.h>
#include <hip/hip_bf16.h>

// ---------------- constants (match reference) ----------------
static constexpr int B_  = 32;
static constexpr int S_  = 512;
static constexpr int D_  = 512;
static constexpr int H_  = 8;
static constexpr int L_  = 6;
static constexpr int NQ_ = 8;
static constexpr int FF_ = 2048;
static constexpr int C_  = 10;
static constexpr int DK_ = 64;        // D/H
static constexpr int BS_ = B_ * S_;   // 16384 rows

typedef _Float16 h16_t;
typedef _Float16 v16h __attribute__((ext_vector_type(16)));
typedef _Float16 v8h  __attribute__((ext_vector_type(8)));
typedef float    v8f  __attribute__((ext_vector_type(8)));

__device__ __forceinline__ v16h pack16(v8h lo, v8h hi) {
  v16h r;
#pragma unroll
  for (int j = 0; j < 8; ++j) { r[j] = lo[j]; r[8 + j] = hi[j]; }
  return r;
}

__device__ __forceinline__ v8f wmma_f16(v16h a, v16h b, v8f c) {
  return __builtin_amdgcn_wmma_f32_16x16x32_f16(false, a, false, b, (short)0, c,
                                                false, false);
}

// CDNA5 async copy global->LDS (ASYNCcnt-tracked). LDS address = low 32 bits of
// the generic __shared__ pointer (flat LDS aperture truncates to offset).
__device__ __forceinline__ void async_copy_b128(const h16_t* gsrc, h16_t* ldst) {
  unsigned loff = (unsigned)(unsigned long long)(void*)ldst;
  asm volatile("global_load_async_to_lds_b128 %0, %1, off"
               :
               : "v"(loff), "v"(gsrc)
               : "memory");
}

// ---------------- embedding + positional encoding ----------------
__global__ void k_embed(const int* __restrict__ x, const float* __restrict__ emb,
                        float* __restrict__ h, h16_t* __restrict__ h16) {
  int idx = blockIdx.x * blockDim.x + threadIdx.x;       // over B*S*D
  int d = idx % D_;
  int bs = idx / D_;
  int s = bs % S_;
  int tok = x[bs];
  float freq = __expf(-(float)(2 * (d >> 1)) * (9.2103403719761836f / (float)D_));
  float pe = (d & 1) ? __cosf((float)s * freq) : __sinf((float)s * freq);
  float v = emb[(size_t)tok * D_ + d] + pe;
  h[idx] = v;
  h16[idx] = (h16_t)v;
}

// ---------------- fp32 -> f16 cast (plain) ----------------
__global__ void k_cast_f16(const float* __restrict__ src, h16_t* __restrict__ dst,
                           int n) {
  int i = blockIdx.x * blockDim.x + threadIdx.x;
  if (i < n) dst[i] = (h16_t)src[i];
}

// ---------------- fp32 -> f16 cast + transpose per layer: dst[l][n][k] = src[l][k][n] ----
__global__ void k_cast_transpose_f16(const float* __restrict__ src,
                                     h16_t* __restrict__ dst, int K, int N,
                                     int total) {
  int i = blockIdx.x * blockDim.x + threadIdx.x;   // over L*K*N, dst-linear
  if (i >= total) return;
  int kn = K * N;
  int l = i / kn;
  int rem = i - l * kn;
  int n = rem / K;
  int k = rem - n * K;
  dst[i] = (h16_t)src[(size_t)l * kn + (size_t)k * N + n];
}

// ---------------- fused flash attention ----------------
// q=k=v = h16 viewed as [B,S,H,DK]. One block = (b, head, 128-row query tile).
// 8 waves; wave owns 16 query rows, accumulates O[16x64] in 4 f32 C-frags.
__global__ __launch_bounds__(256) void k_attn(const h16_t* __restrict__ h16,
                                              h16_t* __restrict__ attn_out) {
  int bid = blockIdx.x;                 // b*H*4 + head*4 + rowtile
  int rt = bid & 3;
  int head = (bid >> 2) & (H_ - 1);
  int b = bid >> 5;
  int wave = threadIdx.x >> 5;
  int lane = threadIdx.x & 31;
  int halflane = lane & 15;
  int hi = lane >> 4;

  const h16_t* base = h16 + (size_t)b * S_ * D_ + head * DK_;
  int qrow0 = rt * 128 + wave * 16;

  // Q A-frags: 2 chained (dk 0..31, 32..63), invariant over key loop
  v16h qf[2];
  {
    const h16_t* qp = base + (size_t)(qrow0 + halflane) * D_;
    int kb = hi * 16;
#pragma unroll
    for (int c = 0; c < 2; ++c) {
      v8h a0 = *(const v8h*)(qp + c * 32 + kb);
      v8h a1 = *(const v8h*)(qp + c * 32 + kb + 8);
      qf[c] = pack16(a0, a1);
    }
  }

  v8f o[4] = {};
  float mrow[8], lrow[8];
#pragma unroll
  for (int r = 0; r < 8; ++r) { mrow[r] = -3.0e38f; lrow[r] = 0.f; }

  __shared__ __align__(16) h16_t VT[64][40];       // V tile transposed [d][k]
  __shared__ __align__(16) h16_t Pb[8][16][40];    // per-wave P staging

  for (int kt = 0; kt < S_ / 32; ++kt) {
    int k0 = kt * 32;
    __syncthreads();   // previous-iteration VT reads done
    {  // cooperative transposed stage of V tile (32 key rows x 64 dims)
      int kr = threadIdx.x >> 3;
      int dcb = (threadIdx.x & 7) * 8;
      v8h v = *(const v8h*)(base + (size_t)(k0 + kr) * D_ + dcb);
#pragma unroll
      for (int j = 0; j < 8; ++j) VT[dcb + j][kr] = v[j];
    }

    // scores S[16x32] = Q x K^T, two col groups x two chained dk halves
    v8f sc[2] = {};
#pragma unroll
    for (int cg = 0; cg < 2; ++cg) {
#pragma unroll
      for (int c = 0; c < 2; ++c) {
        int keypos = k0 + cg * 16 + halflane;
        const h16_t* kp = base + (size_t)keypos * D_ + c * 32 + hi * 16;
        v16h bf = pack16(*(const v8h*)kp, *(const v8h*)(kp + 8));
        sc[cg] = wmma_f16(qf[c], bf, sc[cg]);
      }
    }

    // online softmax; stage P (f16) to LDS in matrix layout
    float corr[8];
#pragma unroll
    for (int r = 0; r < 8; ++r) {
      float s0 = sc[0][r] * 0.125f;
      float s1 = sc[1][r] * 0.125f;
      float mx = fmaxf(s0, s1);
#pragma unroll
      for (int off = 1; off < 16; off <<= 1) mx = fmaxf(mx, __shfl_xor(mx, off, 32));
      float mnew = fmaxf(mrow[r], mx);
      corr[r] = __expf(mrow[r] - mnew);
      float p0 = __expf(s0 - mnew);
      float p1 = __expf(s1 - mnew);
      float ps = p0 + p1;
#pragma unroll
      for (int off = 1; off < 16; off <<= 1) ps += __shfl_xor(ps, off, 32);
      lrow[r] = lrow[r] * corr[r] + ps;
      mrow[r] = mnew;
      int prow = hi * 8 + r;
      Pb[wave][prow][halflane] = (h16_t)p0;
      Pb[wave][prow][16 + halflane] = (h16_t)p1;
    }
    __syncthreads();   // VT staged; P visible

    // O = O*corr + P x V
    v16h pf;
    {
      int kb = hi * 16;
      v8h p0 = *(const v8h*)&Pb[wave][halflane][kb];
      v8h p1 = *(const v8h*)&Pb[wave][halflane][kb + 8];
      pf = pack16(p0, p1);
    }
#pragma unroll
    for (int ng = 0; ng < 4; ++ng) {
      v8f oacc;
#pragma unroll
      for (int r = 0; r < 8; ++r) oacc[r] = o[ng][r] * corr[r];
      int dcol = ng * 16 + halflane;
      v16h bf = pack16(*(const v8h*)&VT[dcol][hi * 16],
                       *(const v8h*)&VT[dcol][hi * 16 + 8]);
      o[ng] = wmma_f16(pf, bf, oacc);
    }
  }

  // finalize and store f16 attention output [B,S,D]
#pragma unroll
  for (int ng = 0; ng < 4; ++ng) {
    int dcol = head * DK_ + ng * 16 + halflane;
#pragma unroll
    for (int r = 0; r < 8; ++r) {
      int row = qrow0 + hi * 8 + r;
      attn_out[(size_t)b * S_ * D_ + (size_t)row * D_ + dcol] =
          (h16_t)(o[ng][r] / lrow[r]);
    }
  }
}

// ---------------- WMMA GEMM: C = A(f16)[MxK] * BT(f16)[NxK]^T + bias ----------------
// 256 threads / 8 waves. Block tile 256x64; wave tile 32x64 (2 A-frags x 4 B-frags,
// 8 WMMAs per 32-K step). Tiles double-buffered in LDS, filled with CDNA5
// global_load_async_to_lds_b128 (ASYNCcnt), s_wait_asynccnt drains prior tile.
static constexpr int BM = 256, BN = 64, BK = 32;
__global__ __launch_bounds__(256) void k_gemm_f16(const h16_t* __restrict__ A,
                                                  const h16_t* __restrict__ BT,
                                                  const float* __restrict__ bias,
                                                  float* __restrict__ Cout,
                                                  int M, int N, int K) {
  __shared__ __align__(16) h16_t As[2][BM][BK + 8];   // +8 halves pad, rows stay 16B aligned
  __shared__ __align__(16) h16_t Bs[2][BN][BK + 8];
  int tid = threadIdx.x;
  int wave = tid >> 5;
  int lane = tid & 31;
  int halflane = lane & 15;
  int hi = lane >> 4;
  int bm = blockIdx.x * BM;
  int bn = blockIdx.y * BN;

  const h16_t* Ag = A + (size_t)(bm + tid) * K;                       // one A row per thread
  const h16_t* Bg = BT + (size_t)(bn + (tid >> 2)) * K + (tid & 3) * 8;
  int brow = tid >> 2;
  int bcb = (tid & 3) * 8;

  v8f acc[2][4] = {};

  // prologue: stage tile 0 (5 async b128 per thread)
#pragma unroll
  for (int j = 0; j < 4; ++j) async_copy_b128(Ag + j * 8, &As[0][tid][j * 8]);
  async_copy_b128(Bg, &Bs[0][brow][bcb]);

  int T = K / BK;
  for (int t = 0; t < T; ++t) {
    int buf = t & 1;
    if (t + 1 < T) {
      int k0 = (t + 1) * BK;
#pragma unroll
      for (int j = 0; j < 4; ++j)
        async_copy_b128(Ag + k0 + j * 8, &As[buf ^ 1][tid][j * 8]);
      async_copy_b128(Bg + k0, &Bs[buf ^ 1][brow][bcb]);
      asm volatile("s_wait_asynccnt 0x5" ::: "memory");   // tile t landed (in-order)
    } else {
      asm volatile("s_wait_asynccnt 0x0" ::: "memory");
    }
    __syncthreads();

    int kb = hi * 16;
    int kbB = hi * 16;
    v16h af[2];
#pragma unroll
    for (int i = 0; i < 2; ++i) {
      int fr = wave * 32 + i * 16 + halflane;
      af[i] = pack16(*(const v8h*)&As[buf][fr][kb], *(const v8h*)&As[buf][fr][kb + 8]);
    }
#pragma unroll
    for (int ng = 0; ng < 4; ++ng) {
      int col = ng * 16 + halflane;
      v16h bf = pack16(*(const v8h*)&Bs[buf][col][kbB], *(const v8h*)&Bs[buf][col][kbB + 8]);
#pragma unroll
      for (int i = 0; i < 2; ++i) acc[i][ng] = wmma_f16(af[i], bf, acc[i][ng]);
    }
    __syncthreads();
  }

#pragma unroll
  for (int i = 0; i < 2; ++i)
#pragma unroll
    for (int ng = 0; ng < 4; ++ng) {
      int col = bn + ng * 16 + halflane;
      float bv = bias[col];
#pragma unroll
      for (int r = 0; r < 8; ++r) {
        int row = bm + wave * 32 + i * 16 + hi * 8 + r;
        Cout[(size_t)row * N + col] = acc[i][ng][r] + bv;
      }
    }
}

// ---------------- residual + layernorm (writes fp32 h and f16 shadow) ----------------
__global__ __launch_bounds__(256) void k_ln_res(float* __restrict__ h,
                                                const float* __restrict__ y,
                                                const float* __restrict__ g,
                                                const float* __restrict__ beta,
                                                h16_t* __restrict__ h16) {
  int row = blockIdx.x;                     // B*S rows
  int tid = threadIdx.x;
  size_t base = (size_t)row * D_;
  float x0 = h[base + tid] + y[base + tid];
  float x1 = h[base + tid + 256] + y[base + tid + 256];
  float s = x0 + x1;
  float q = x0 * x0 + x1 * x1;
#pragma unroll
  for (int off = 1; off < 32; off <<= 1) {
    s += __shfl_xor(s, off, 32);
    q += __shfl_xor(q, off, 32);
  }
  __shared__ float rs[8], rq[8];
  if ((tid & 31) == 0) { rs[tid >> 5] = s; rq[tid >> 5] = q; }
  __syncthreads();
  float ts = 0.f, tq = 0.f;
#pragma unroll
  for (int w = 0; w < 8; ++w) { ts += rs[w]; tq += rq[w]; }
  float mu = ts * (1.0f / D_);
  float var = tq * (1.0f / D_) - mu * mu;
  float inv = rsqrtf(var + 1e-5f);
  float o0 = (x0 - mu) * inv * g[tid] + beta[tid];
  float o1 = (x1 - mu) * inv * g[tid + 256] + beta[tid + 256];
  h[base + tid] = o0;
  h[base + tid + 256] = o1;
  h16[base + tid] = (h16_t)o0;
  h16[base + tid + 256] = (h16_t)o1;
}

// ---------------- quantum encode: qout = cos(h[..., :NQ]) * cos(theta) ----------------
__global__ void k_qout(const float* __restrict__ h, const float* __restrict__ theta,
                       float* __restrict__ qout) {
  int i = blockIdx.x * blockDim.x + threadIdx.x;   // B*S*NQ
  int q = i & (NQ_ - 1);
  int row = i >> 3;
  qout[i] = __cosf(h[(size_t)row * D_ + q]) * __cosf(theta[q]);
}

// ---------------- FFN front: f = relu(qout @ W1 + b1), K=8, f16 out ----------------
__global__ __launch_bounds__(256) void k_ffn1(const float* __restrict__ qout,
                                              const float* __restrict__ w1,
                                              const float* __restrict__ b1,
                                              h16_t* __restrict__ f) {
  int idx = blockIdx.x * blockDim.x + threadIdx.x;   // B*S*FF
  int ff = idx & (FF_ - 1);
  int row = idx / FF_;
  float acc = b1[ff];
#pragma unroll
  for (int k = 0; k < NQ_; ++k)
    acc += qout[(size_t)row * NQ_ + k] * w1[(size_t)k * FF_ + ff];
  f[idx] = (h16_t)fmaxf(acc, 0.f);
}

// ---------------- mean pool over S ----------------
__global__ void k_pool(const float* __restrict__ h, float* __restrict__ pooled) {
  int i = blockIdx.x * blockDim.x + threadIdx.x;   // B*D
  int d = i & (D_ - 1);
  int b = i / D_;
  float s = 0.f;
  for (int t = 0; t < S_; ++t) s += h[((size_t)b * S_ + t) * D_ + d];
  pooled[i] = s * (1.0f / S_);
}

// ---------------- classifier ----------------
__global__ void k_clf(const float* __restrict__ pooled, const float* __restrict__ w,
                      const float* __restrict__ bias, float* __restrict__ out) {
  int i = blockIdx.x * blockDim.x + threadIdx.x;   // B*C
  if (i >= B_ * C_) return;
  int c = i % C_;
  int b = i / C_;
  float s = bias[c];
  for (int d = 0; d < D_; ++d) s += pooled[(size_t)b * D_ + d] * w[(size_t)d * C_ + c];
  out[i] = s;
}

// ---------------- host orchestration ----------------
extern "C" void kernel_launch(void* const* d_in, const int* in_sizes, int n_in,
                              void* d_out, int out_size, void* d_ws, size_t ws_size,
                              hipStream_t stream) {
  (void)in_sizes; (void)n_in; (void)out_size; (void)ws_size;
  const int*   x      = (const int*)  d_in[0];
  const float* emb    = (const float*)d_in[1];
  const float* ln1_g  = (const float*)d_in[2];
  const float* ln1_b  = (const float*)d_in[3];
  const float* ln2_g  = (const float*)d_in[4];
  const float* ln2_b  = (const float*)d_in[5];
  const float* attn_w = (const float*)d_in[6];
  const float* attn_b = (const float*)d_in[7];
  const float* theta  = (const float*)d_in[8];
  const float* ffn_w1 = (const float*)d_in[9];
  const float* ffn_b1 = (const float*)d_in[10];
  const float* ffn_w2 = (const float*)d_in[11];
  const float* ffn_b2 = (const float*)d_in[12];
  const float* clf_w  = (const float*)d_in[13];
  const float* clf_b  = (const float*)d_in[14];
  float* out = (float*)d_out;

  char* ws = (char*)d_ws;
  size_t off = 0;
  auto carve = [&](size_t bytes) -> char* {
    char* p = ws + off;
    off = (off + bytes + 255) & ~(size_t)255;
    return p;
  };
  float* h        = (float*)carve((size_t)BS_ * D_ * 4);
  h16_t* h16      = (h16_t*)carve((size_t)BS_ * D_ * 2);
  float* y        = (float*)carve((size_t)BS_ * D_ * 4);
  h16_t* attn16   = (h16_t*)carve((size_t)BS_ * D_ * 2);
  h16_t* fbuf     = (h16_t*)carve((size_t)BS_ * FF_ * 2);
  float* qoutbuf  = (float*)carve((size_t)BS_ * NQ_ * 4);
  h16_t* attnw16T = (h16_t*)carve((size_t)L_ * D_ * D_ * 2);    // [l][N][K]
  h16_t* w2_16T   = (h16_t*)carve((size_t)L_ * FF_ * D_ * 2);   // [l][N=512][K=2048]
  float* pooled   = (float*)carve((size_t)B_ * D_ * 4);

  // weight cast + transpose (deterministic each call)
  k_cast_transpose_f16<<<(L_ * D_ * D_) / 256, 256, 0, stream>>>(
      attn_w, attnw16T, D_, D_, L_ * D_ * D_);
  k_cast_transpose_f16<<<(L_ * FF_ * D_) / 256, 256, 0, stream>>>(
      ffn_w2, w2_16T, FF_, D_, L_ * FF_ * D_);

  k_embed<<<(BS_ * D_) / 256, 256, 0, stream>>>(x, emb, h, h16);

  dim3 gemm_grid(BS_ / BM, D_ / BN);
  for (int l = 0; l < L_; ++l) {
    k_attn<<<B_ * H_ * 4, 256, 0, stream>>>(h16, attn16);
    k_gemm_f16<<<gemm_grid, 256, 0, stream>>>(attn16, attnw16T + (size_t)l * D_ * D_,
                                              attn_b + (size_t)l * D_, y, BS_, D_, D_);
    k_ln_res<<<BS_, 256, 0, stream>>>(h, y, ln1_g + (size_t)l * D_,
                                      ln1_b + (size_t)l * D_, h16);
    k_qout<<<(BS_ * NQ_) / 256, 256, 0, stream>>>(h, theta + (size_t)l * NQ_, qoutbuf);
    k_ffn1<<<(BS_ * FF_) / 256, 256, 0, stream>>>(qoutbuf, ffn_w1 + (size_t)l * NQ_ * FF_,
                                                  ffn_b1 + (size_t)l * FF_, fbuf);
    k_gemm_f16<<<gemm_grid, 256, 0, stream>>>(fbuf, w2_16T + (size_t)l * FF_ * D_,
                                              ffn_b2 + (size_t)l * D_, y, BS_, D_, FF_);
    k_ln_res<<<BS_, 256, 0, stream>>>(h, y, ln2_g + (size_t)l * D_,
                                      ln2_b + (size_t)l * D_, h16);
  }

  k_pool<<<(B_ * D_) / 256, 256, 0, stream>>>(h, pooled);
  k_clf<<<(B_ * C_ + 255) / 256, 256, 0, stream>>>(pooled, clf_w, clf_b, out);
}
